// WGNN_44074954391863
// MI455X (gfx1250) — matrix operand order repaired and checked by reading, not compiled, not run on hardware
//
#include <hip/hip_runtime.h>

typedef __attribute__((ext_vector_type(16))) __bf16 v16bf;
typedef __attribute__((ext_vector_type(8)))  float  v8f;

#define DCOL 256
#define HCOL 128
#define BM 128
#define BN 64
#define AS 264   // padded LDS row stride (bf16 elems)
#define BS 264

__device__ __forceinline__ unsigned short f2bf(float f) {
  union { float f; unsigned u; } c; c.f = f;
  unsigned u = c.u;
  u += 0x7FFFu + ((u >> 16) & 1u);          // round-to-nearest-even
  return (unsigned short)(u >> 16);
}

// ---------------------------------------------------------------- init / util
__global__ void initz_kernel(const float* __restrict__ x, float* __restrict__ z, int n) {
  size_t tid = blockIdx.x * (size_t)blockDim.x + threadIdx.x;
  size_t tot = (size_t)n * 64;                  // 64 float4 per 256-col row
  if (tid >= tot) return;
  int row = (int)(tid >> 6);
  int c4  = (int)(tid & 63);
  float4 v;
  if (c4 < 32) v = *(const float4*)(x + (size_t)row * HCOL + c4 * 4);
  else         v = make_float4(0.f, 0.f, 0.f, 0.f);
  ((float4*)z)[tid] = v;
}

__global__ void copy_alpha_kernel(const float* __restrict__ a0, float* __restrict__ alpha, int n) {
  int i = blockIdx.x * blockDim.x + threadIdx.x;
  if (i < n) alpha[i] = a0[i];
}

__global__ void zero_kernel(float4* __restrict__ p, size_t n4) {
  size_t i = blockIdx.x * (size_t)blockDim.x + threadIdx.x;
  if (i < n4) p[i] = make_float4(0.f, 0.f, 0.f, 0.f);
}

__global__ void extract_kernel(const float* __restrict__ z, float* __restrict__ out, int n) {
  size_t tid = blockIdx.x * (size_t)blockDim.x + threadIdx.x;
  size_t tot = (size_t)n * 32;                  // first 128 cols = 32 float4 per row
  if (tid >= tot) return;
  int row = (int)(tid >> 5);
  int c4  = (int)(tid & 31);
  ((float4*)out)[tid] = ((const float4*)z)[(size_t)row * 64 + c4];
}

// w = (W * clip(d,0,1)) @ W^T  (symmetric), stored directly as bf16
__global__ void wcompute_kernel(const float* __restrict__ W, const float* __restrict__ dvec,
                                unsigned short* __restrict__ wbf) {
  int i = blockIdx.x;
  int j = threadIdx.x;
  float s = 0.f;
  for (int k = 0; k < DCOL; ++k) {
    float d = dvec[k];
    d = d < 0.f ? 0.f : (d > 1.f ? 1.f : d);
    s += W[(size_t)i * DCOL + k] * d * W[(size_t)j * DCOL + k];
  }
  wbf[(size_t)i * DCOL + j] = f2bf(s);
}

// ------------------------------------------------- alpha / sigmoid (1 wave/node)
__global__ void alpha_kernel(const float* __restrict__ zeval, const float* __restrict__ h,
                             const float* __restrict__ W_ih, const float* __restrict__ W_hh,
                             const float* __restrict__ b_ih, const float* __restrict__ b_hh,
                             float* __restrict__ alpha, float* __restrict__ alph, int n) {
  int gw   = (int)((blockIdx.x * (size_t)blockDim.x + threadIdx.x) >> 5);
  int lane = threadIdx.x & 31;
  if (gw >= n) return;
  const float* zr = zeval + (size_t)gw * DCOL;
  float s0 = 0.f, s1 = 0.f;
#pragma unroll
  for (int j = 0; j < 8; ++j) {
    int k = lane + j * 32;
    float z = zr[k];
    s0 += z * W_ih[k];
    s1 += z * W_ih[DCOL + k];
  }
#pragma unroll
  for (int off = 16; off >= 1; off >>= 1) {
    s0 += __shfl_xor(s0, off, 32);
    s1 += __shfl_xor(s1, off, 32);
  }
  if (lane == 0) {
    float h0 = h[(size_t)gw * 2], h1 = h[(size_t)gw * 2 + 1];
    float a0 = tanhf(s0 + b_ih[0] + h0 * W_hh[0] + h1 * W_hh[1] + b_hh[0]);
    float a1 = tanhf(s1 + b_ih[1] + h0 * W_hh[2] + h1 * W_hh[3] + b_hh[1]);
    float an = alpha[gw] * a0 + a1;
    alpha[gw] = an;
    alph[gw]  = 1.f / (1.f + expf(-an));
  }
}

// ------------------------------------------------- SpMM: az[row] += z[col]*val
__global__ void spmm_kernel(const float* __restrict__ zeval, const int* __restrict__ er,
                            const int* __restrict__ ec, const float* __restrict__ ev,
                            float* __restrict__ az, int e) {
  int gw   = (int)((blockIdx.x * (size_t)blockDim.x + threadIdx.x) >> 5);
  int lane = threadIdx.x & 31;
  if (gw >= e) return;
  int r = er[gw], c = ec[gw];
  float v = ev[gw];
  const float4* src = (const float4*)(zeval + (size_t)c * DCOL);
  float* dst = az + (size_t)r * DCOL;
#pragma unroll
  for (int half = 0; half < 2; ++half) {
    int i = lane + half * 32;
    float4 p = src[i];
    atomicAdd(dst + i * 4 + 0, p.x * v);
    atomicAdd(dst + i * 4 + 1, p.y * v);
    atomicAdd(dst + i * 4 + 2, p.z * v);
    atomicAdd(dst + i * 4 + 3, p.w * v);
  }
}

// ---------------- fused WMMA GEMM (z@w) + f-epilogue + RK4 stage combination
// f = acc + alph*0.5*(az - z) - z + x0
// stage<4:  zsum' = (stage==1 ? zbase : zsum) + wc*f ;  zout = zbase + cc*f
// stage==4: zbase = zsum + wc*f
__global__ __launch_bounds__(256) void gemm_stage_kernel(
    const float* __restrict__ zeval, const float* __restrict__ az,
    const float* __restrict__ x, const float* __restrict__ alph,
    const unsigned short* __restrict__ wbf,
    const float* __restrict__ zbase, float* __restrict__ zbase_w,
    float* __restrict__ zsum, float* __restrict__ zout,
    int n, int stage, float cc, float wc) {
  __shared__ __align__(16) unsigned short Al[BM * AS];
  __shared__ __align__(16) unsigned short Bl[BN * BS];
  int t  = threadIdx.x;
  int m0 = blockIdx.x * BM;
  int n0 = blockIdx.y * BN;

  // A tile: 128x256 f32 -> bf16 in LDS (row i fully coalesced per iteration)
#pragma unroll 4
  for (int it = 0; it < 32; ++it) {
    int f4  = t + it * 256;          // 8192 float4 total
    int row = f4 >> 6;
    int c4  = f4 & 63;
    int gr  = m0 + row;
    if (gr >= n) gr = n - 1;         // clamp; tail rows masked at store time
    float4 p = *(const float4*)(zeval + (size_t)gr * DCOL + c4 * 4);
    unsigned u0 = (unsigned)f2bf(p.x) | ((unsigned)f2bf(p.y) << 16);
    unsigned u1 = (unsigned)f2bf(p.z) | ((unsigned)f2bf(p.w) << 16);
    int off = row * AS + c4 * 4;
    *(unsigned*)&Al[off]     = u0;
    *(unsigned*)&Al[off + 2] = u1;
  }
  // B tile: rows n0..n0+63 of symmetric w (already bf16)
#pragma unroll
  for (int it = 0; it < 8; ++it) {
    int q  = t + it * 256;           // 2048 chunks of 8 ushorts
    int br = q >> 5;
    int c8 = q & 31;
    *(uint4*)&Bl[br * BS + c8 * 8] =
        *(const uint4*)(wbf + (size_t)(n0 + br) * DCOL + c8 * 8);
  }
  __syncthreads();

  int wave = t >> 5;
  int lane = t & 31;
  int lr = lane & 15, lh = lane >> 4;
  int arow = wave * 16 + lr;

  union FragBF { uint4 u[2]; v16bf bf; };
  v8f acc[4] = {};

#pragma unroll
  for (int kk = 0; kk < DCOL; kk += 32) {
    FragBF a;
    a.u[0] = *(const uint4*)&Al[arow * AS + kk + lh * 8];
    a.u[1] = *(const uint4*)&Al[arow * AS + kk + lh * 8 + 16];
#pragma unroll
    for (int ct = 0; ct < 4; ++ct) {
      FragBF b;
      int brow = ct * 16 + lr;
      b.u[0] = *(const uint4*)&Bl[brow * BS + kk + lh * 8];
      b.u[1] = *(const uint4*)&Bl[brow * BS + kk + lh * 8 + 16];
      acc[ct] = __builtin_amdgcn_wmma_f32_16x16x32_bf16(
          false, a.bf, false, b.bf, (short)0, acc[ct], false, false);
    }
  }

  // epilogue: C layout -> element (M = v + 8*lh, N = lr) within 16x16 fragment
#pragma unroll
  for (int ct = 0; ct < 4; ++ct) {
    int ng = n0 + ct * 16 + lr;
#pragma unroll
    for (int v = 0; v < 8; ++v) {
      int mg = m0 + wave * 16 + v + lh * 8;
      if (mg < n) {
        size_t idx = (size_t)mg * DCOL + ng;
        float zv  = zeval[idx];
        float azv = az[idx];
        float x0v = (ng < HCOL) ? x[(size_t)mg * HCOL + ng] : 0.f;
        float al  = alph[mg];
        float f   = acc[ct][v] + al * 0.5f * (azv - zv) - zv + x0v;
        float zs  = (stage == 1) ? zbase[idx] : zsum[idx];
        float zsn = zs + wc * f;
        if (stage < 4) {
          zsum[idx] = zsn;
          zout[idx] = zbase[idx] + cc * f;
        } else {
          zbase_w[idx] = zsn;
        }
      }
    }
  }
}

// --------------------------------------------------------------------- driver
extern "C" void kernel_launch(void* const* d_in, const int* in_sizes, int n_in,
                              void* d_out, int out_size, void* d_ws, size_t ws_size,
                              hipStream_t stream) {
  (void)n_in; (void)out_size; (void)ws_size;
  const float* x      = (const float*)d_in[0];
  const int*   er     = (const int*)d_in[1];
  const int*   ec     = (const int*)d_in[2];
  const float* ev     = (const float*)d_in[3];
  const float* W_ih   = (const float*)d_in[4];
  const float* W_hh   = (const float*)d_in[5];
  const float* b_ih   = (const float*)d_in[6];
  const float* b_hh   = (const float*)d_in[7];
  const float* h      = (const float*)d_in[8];
  const float* alpha0 = (const float*)d_in[9];
  const float* W      = (const float*)d_in[10];
  const float* dvec   = (const float*)d_in[11];

  const int n = in_sizes[9];   // N nodes (alpha0 length)
  const int e = in_sizes[1];   // E edges

  size_t szZ = (size_t)n * DCOL * sizeof(float);
  char* ws = (char*)d_ws;
  float* z      = (float*)ws;
  float* zsum   = (float*)(ws + 1 * szZ);
  float* zevalA = (float*)(ws + 2 * szZ);
  float* zevalB = (float*)(ws + 3 * szZ);
  float* az     = (float*)(ws + 4 * szZ);
  float* alpha  = (float*)(ws + 5 * szZ);
  float* alph   = alpha + n;
  size_t woff = ((5 * szZ + 2 * (size_t)n * sizeof(float)) + 255) & ~(size_t)255;
  unsigned short* wbf = (unsigned short*)(ws + woff);

  dim3 b256(256);
  size_t nz4 = (size_t)n * 64;
  initz_kernel<<<dim3((unsigned)((nz4 + 255) / 256)), b256, 0, stream>>>(x, z, n);
  copy_alpha_kernel<<<dim3((n + 255) / 256), b256, 0, stream>>>(alpha0, alpha, n);
  wcompute_kernel<<<dim3(DCOL), b256, 0, stream>>>(W, dvec, wbf);

  const float dt = 0.9f / 2.0f;
  const float ccs[4] = {0.5f * dt, 0.5f * dt, dt, 0.f};
  const float wcs[4] = {dt / 6.f, dt / 3.f, dt / 3.f, dt / 6.f};

  dim3 ggrid((n + BM - 1) / BM, DCOL / BN);

  for (int step = 0; step < 2; ++step) {
    for (int st = 1; st <= 4; ++st) {
      const float* zeval = (st == 1) ? z : (st == 2 ? zevalA : (st == 3 ? zevalB : zevalA));
      float* zout        = (st == 1) ? zevalA : (st == 2 ? zevalB : (st == 3 ? zevalA : nullptr));
      alpha_kernel<<<dim3((n + 7) / 8), b256, 0, stream>>>(zeval, h, W_ih, W_hh,
                                                           b_ih, b_hh, alpha, alph, n);
      zero_kernel<<<dim3((unsigned)((nz4 + 255) / 256)), b256, 0, stream>>>((float4*)az, nz4);
      spmm_kernel<<<dim3((e + 7) / 8), b256, 0, stream>>>(zeval, er, ec, ev, az, e);
      gemm_stage_kernel<<<ggrid, b256, 0, stream>>>(zeval, az, x, alph, wbf,
                                                    z, z, zsum, zout,
                                                    n, st, ccs[st - 1], wcs[st - 1]);
    }
  }

  size_t no4 = (size_t)n * 32;
  extract_kernel<<<dim3((unsigned)((no4 + 255) / 256)), b256, 0, stream>>>(z, (float*)d_out, n);
}
